// Decoder_33672543600930
// MI455X (gfx1250) — matrix operand order repaired and checked by reading, not compiled, not run on hardware
//
#include <hip/hip_runtime.h>
#include <hip/hip_bf16.h>

// ---------------------------------------------------------------------------
// GRU decoder for MI455X (gfx1250, wave32, WMMA bf16).
// B=256, T=256, I=256, H=1024, L=128.
// Strategy: convert weights to bf16 once (L2-resident); all GEMMs via
// v_wmma_f32_16x16x32_bf16 with f32 accumulation. One fused kernel per
// timestep: blocks 0-15 compute gates + h_t, blocks 16-19 concurrently
// compute y_{t-1} (output GEMM overlaps the recurrent GEMM).
// ---------------------------------------------------------------------------

#define BB 256
#define TT 256
#define II 256
#define HH 1024
#define LL 128

typedef __bf16 v16bf __attribute__((ext_vector_type(16)));
typedef __bf16 bf16x8 __attribute__((ext_vector_type(8)));
typedef float  v8f    __attribute__((ext_vector_type(8)));
typedef float  f32x4  __attribute__((ext_vector_type(4)));

static __device__ __forceinline__ v8f wmma_bf16(v16bf a, v16bf b, v8f c) {
    // (neg_a, A, neg_b, B, c_mod, C, reuse_a, reuse_b)
    return __builtin_amdgcn_wmma_f32_16x16x32_bf16(false, a, false, b,
                                                   (short)0, c, false, false);
}

// A fragment (16x32 bf16, MxK): lane<16 -> row M=lane, K 0..15;
// lane>=16 -> row M=lane-16, K 16..31. `p` already includes
// row*ld + k0 + (laneHi ? 16 : 0).
static __device__ __forceinline__ v16bf load_a_bf(const __bf16* p) {
    bf16x8 lo = *(const bf16x8*)(p);
    bf16x8 hi = *(const bf16x8*)(p + 8);
    return __builtin_shufflevector(lo, hi, 0, 1, 2, 3, 4, 5, 6, 7,
                                   8, 9, 10, 11, 12, 13, 14, 15);
}

// Same A fragment but sourced from f32 memory, converted to bf16 on the fly.
static __device__ __forceinline__ v16bf load_a_f32(const float* p) {
    f32x4 a0 = *(const f32x4*)(p);
    f32x4 a1 = *(const f32x4*)(p + 4);
    f32x4 a2 = *(const f32x4*)(p + 8);
    f32x4 a3 = *(const f32x4*)(p + 12);
    v16bf r;
#pragma unroll
    for (int i = 0; i < 4; ++i) {
        r[i]      = (__bf16)a0[i];
        r[4 + i]  = (__bf16)a1[i];
        r[8 + i]  = (__bf16)a2[i];
        r[12 + i] = (__bf16)a3[i];
    }
    return r;
}

// B fragment (32x16 bf16, KxN): lane holds column N=lane%16, 16 contiguous K
// values at k0 + 16*(lane/16). `p` already includes col*ld + k0 + laneHi*16.
static __device__ __forceinline__ v16bf load_b_bf(const __bf16* p) {
    bf16x8 lo = *(const bf16x8*)(p);
    bf16x8 hi = *(const bf16x8*)(p + 8);
    return __builtin_shufflevector(lo, hi, 0, 1, 2, 3, 4, 5, 6, 7,
                                   8, 9, 10, 11, 12, 13, 14, 15);
}

static __device__ __forceinline__ float sigmoidf_(float x) {
    return 1.0f / (1.0f + __expf(-x));
}

// ---------------------------------------------------------------------------
// One-time: convert W_ih [3H,I], W_hh [3H,H], W_out [I,H] to bf16.
// ---------------------------------------------------------------------------
#define NIH (3 * HH * II)
#define NHH (3 * HH * HH)
#define NWO (II * HH)

__global__ __launch_bounds__(256) void cvt_weights_kernel(
    const float* __restrict__ wih, const float* __restrict__ whh,
    const float* __restrict__ wout, __bf16* __restrict__ wihb,
    __bf16* __restrict__ whhb, __bf16* __restrict__ woutb) {
    int idx = blockIdx.x * 256 + threadIdx.x;
    if (idx < NIH) {
        wihb[idx] = (__bf16)wih[idx];
    } else if (idx < NIH + NHH) {
        int k = idx - NIH;
        whhb[k] = (__bf16)whh[k];
    } else {
        int k = idx - NIH - NHH;
        woutb[k] = (__bf16)wout[k];
    }
}

// ---------------------------------------------------------------------------
// h0 = tanh(z @ W_z2h^T + b_z2h); write both f32 and bf16 copies.
// ---------------------------------------------------------------------------
__global__ __launch_bounds__(256) void h0_kernel(
    const float* __restrict__ z, const float* __restrict__ wz2h,
    const float* __restrict__ bz2h, float* __restrict__ hf,
    __bf16* __restrict__ hb) {
    int idx = blockIdx.x * 256 + threadIdx.x;   // b*H + h
    int b = idx >> 10;
    int h = idx & (HH - 1);
    const float* zr = z + (size_t)b * LL;
    const float* wr = wz2h + (size_t)h * LL;
    float acc = bz2h[h];
#pragma unroll 4
    for (int l = 0; l < LL; ++l) acc += zr[l] * wr[l];
    float v = tanhf(acc);
    hf[idx] = v;
    hb[idx] = (__bf16)v;
}

// ---------------------------------------------------------------------------
// Output GEMM body: y_t = sigmoid(h @ W_out^T + b_out) -> out[:, t, :]
// (16x16 tile per wave; caller supplies tile bases).
// ---------------------------------------------------------------------------
static __device__ __forceinline__ void out_tile(
    const __bf16* __restrict__ hb, const __bf16* __restrict__ Wout,
    const float* __restrict__ bout, float* __restrict__ out, int t, int m0,
    int n0, int lm, int laneHi) {
    const int arow = m0 + lm;
    const int brow = n0 + lm;

    v8f acc = {};
    const __bf16* hbase = hb + (size_t)arow * HH + laneHi * 16;
    const __bf16* wbase = Wout + (size_t)brow * HH + laneHi * 16;
    for (int k = 0; k < HH; k += 32) {
        acc = wmma_bf16(load_a_bf(hbase + k), load_b_bf(wbase + k), acc);
    }

    const int i = n0 + lm;
    const float bo = bout[i];
#pragma unroll
    for (int e = 0; e < 8; ++e) {
        const int row = m0 + laneHi * 8 + e;
        out[(size_t)row * TT * II + (size_t)t * II + i] =
            sigmoidf_(acc[e] + bo);
    }
}

// ---------------------------------------------------------------------------
// Fused step kernel, grid (20, 8), block 256 (8 waves, 2x4 tile grid):
//   blockIdx.x in [0,16):  gate work for step t:
//     acc_r  = x_t@Wr^T + h@Ur^T    acc_z = x_t@Wz^T + h@Uz^T
//     acc_xn = x_t@Wn^T             acc_hn = h@Un^T
//     r = sig(acc_r+bir+bhr); zg = sig(acc_z+biz+bhz)
//     n = tanh(acc_xn+bin + r*(acc_hn+bhn)); h_t = (1-zg)*n + zg*h
//   blockIdx.x in [16,20): y_{t-1} = sigmoid(h_{t-1}@W_out^T + b_out)
//     (reads hb_cur only -> races with nothing; overlaps the gate GEMM)
// ---------------------------------------------------------------------------
__global__ __launch_bounds__(256) void gru_step_kernel(
    const float* __restrict__ target, const __bf16* __restrict__ Wih,
    const __bf16* __restrict__ Whh, const float* __restrict__ bih,
    const float* __restrict__ bhh, const __bf16* __restrict__ hb_cur,
    const float* __restrict__ hf_cur, __bf16* __restrict__ hb_nxt,
    float* __restrict__ hf_nxt, const __bf16* __restrict__ Wout,
    const float* __restrict__ bout, float* __restrict__ out, int t) {
    const int tid = threadIdx.x;
    const int lane = tid & 31;
    const int wave = tid >> 5;
    const int wm = wave >> 2;          // 0..1
    const int wn = wave & 3;           // 0..3
    const int lm = lane & 15;
    const int laneHi = lane >> 4;
    const int m0 = blockIdx.y * 32 + wm * 16;   // batch row base

    if (blockIdx.x >= 16) {
        // ---- output blocks: emit y_{t-1} while gate blocks compute h_t ----
        if (t == 0) return;
        const int n0 = (blockIdx.x - 16) * 64 + wn * 16;   // I-column base
        out_tile(hb_cur, Wout, bout, out, t - 1, m0, n0, lm, laneHi);
        return;
    }

    // ---- gate blocks ----
    const int n0 = blockIdx.x * 64 + wn * 16;   // H-column base
    const int arow = m0 + lm;                   // A-fragment source row (b)
    const int brow = n0 + lm;                   // B-fragment source row (n)

    v8f accR = {}, accZ = {}, accXN = {}, accHN = {};

    // input side: K = I = 256, x_t is f32 (convert on the fly)
    const float* xbase =
        target + ((size_t)arow * TT + t) * II + laneHi * 16;
    const __bf16* wr = Wih + (size_t)(0 * HH + brow) * II + laneHi * 16;
    const __bf16* wz = Wih + (size_t)(1 * HH + brow) * II + laneHi * 16;
    const __bf16* wn_ = Wih + (size_t)(2 * HH + brow) * II + laneHi * 16;
    for (int k = 0; k < II; k += 32) {
        v16bf a = load_a_f32(xbase + k);
        accR = wmma_bf16(a, load_b_bf(wr + k), accR);
        accZ = wmma_bf16(a, load_b_bf(wz + k), accZ);
        accXN = wmma_bf16(a, load_b_bf(wn_ + k), accXN);
    }

    // hidden side: K = H = 1024, h is bf16
    const __bf16* hbase = hb_cur + (size_t)arow * HH + laneHi * 16;
    const __bf16* ur = Whh + (size_t)(0 * HH + brow) * HH + laneHi * 16;
    const __bf16* uz = Whh + (size_t)(1 * HH + brow) * HH + laneHi * 16;
    const __bf16* un = Whh + (size_t)(2 * HH + brow) * HH + laneHi * 16;
    for (int k = 0; k < HH; k += 32) {
        v16bf a = load_a_bf(hbase + k);
        accR = wmma_bf16(a, load_b_bf(ur + k), accR);
        accZ = wmma_bf16(a, load_b_bf(uz + k), accZ);
        accHN = wmma_bf16(a, load_b_bf(un + k), accHN);
    }

    // gate math + h update
    const int j = n0 + lm;
    const float br = bih[j] + bhh[j];
    const float bz = bih[HH + j] + bhh[HH + j];
    const float bxn = bih[2 * HH + j];
    const float bhn = bhh[2 * HH + j];

#pragma unroll
    for (int e = 0; e < 8; ++e) {
        // C/D layout: VGPR e holds row M = e (lanes 0-15) / e+8 (lanes 16-31)
        const int row = m0 + laneHi * 8 + e;
        float r = sigmoidf_(accR[e] + br);
        float zg = sigmoidf_(accZ[e] + bz);
        float n = tanhf(accXN[e] + bxn + r * (accHN[e] + bhn));
        float hold = hf_cur[(size_t)row * HH + j];
        float hnew = (1.0f - zg) * n + zg * hold;
        hf_nxt[(size_t)row * HH + j] = hnew;
        hb_nxt[(size_t)row * HH + j] = (__bf16)hnew;
    }
}

// ---------------------------------------------------------------------------
// Epilogue: y_{T-1} from the final hidden state. Grid (4, 8).
// ---------------------------------------------------------------------------
__global__ __launch_bounds__(256) void gru_out_kernel(
    const __bf16* __restrict__ hb, const __bf16* __restrict__ Wout,
    const float* __restrict__ bout, float* __restrict__ out, int t) {
    const int tid = threadIdx.x;
    const int lane = tid & 31;
    const int wave = tid >> 5;
    const int wm = wave >> 2;
    const int wn = wave & 3;
    const int lm = lane & 15;
    const int laneHi = lane >> 4;
    const int m0 = blockIdx.y * 32 + wm * 16;
    const int n0 = blockIdx.x * 64 + wn * 16;
    out_tile(hb, Wout, bout, out, t, m0, n0, lm, laneHi);
}

// ---------------------------------------------------------------------------
// Host side
// ---------------------------------------------------------------------------
extern "C" void kernel_launch(void* const* d_in, const int* in_sizes, int n_in,
                              void* d_out, int out_size, void* d_ws,
                              size_t ws_size, hipStream_t stream) {
    const float* z = (const float*)d_in[0];
    // d_in[1] = seq_len (== TT, hardcoded)
    const float* target = (const float*)d_in[2];
    const float* W_z2h = (const float*)d_in[3];
    const float* b_z2h = (const float*)d_in[4];
    const float* W_ih = (const float*)d_in[5];
    const float* W_hh = (const float*)d_in[6];
    const float* b_ih = (const float*)d_in[7];
    const float* b_hh = (const float*)d_in[8];
    const float* W_out = (const float*)d_in[9];
    const float* b_out = (const float*)d_in[10];
    float* out = (float*)d_out;

    // Workspace layout (bytes), ~11.5 MB total.
    char* ws = (char*)d_ws;
    __bf16* Wih_bf = (__bf16*)(ws + 0);                     // 3072*256 bf16
    __bf16* Whh_bf = (__bf16*)(ws + 1572864);               // 3072*1024 bf16
    __bf16* Wout_bf = (__bf16*)(ws + 7864320);              // 256*1024 bf16
    __bf16* hb[2] = {(__bf16*)(ws + 8388608),               // 256*1024 bf16
                     (__bf16*)(ws + 8912896)};
    float* hf[2] = {(float*)(ws + 9437184),                 // 256*1024 f32
                    (float*)(ws + 10485760)};

    // 1) weights -> bf16 (L2-resident afterwards: ~8.3 MB)
    {
        int n = NIH + NHH + NWO;
        cvt_weights_kernel<<<n / 256, 256, 0, stream>>>(
            W_ih, W_hh, W_out, Wih_bf, Whh_bf, Wout_bf);
    }

    // 2) h0 = tanh(z @ W_z2h^T + b_z2h)
    h0_kernel<<<(BB * HH) / 256, 256, 0, stream>>>(z, W_z2h, b_z2h, hf[0],
                                                   hb[0]);

    // 3) recurrence: 256 fused steps (gate(t) || out(t-1)), graph-captured
    dim3 step_grid(20, BB / 32);        // 16 gate blocks + 4 output blocks
    int cur = 0;
    for (int t = 0; t < TT; ++t) {
        int nxt = cur ^ 1;
        gru_step_kernel<<<step_grid, 256, 0, stream>>>(
            target, Wih_bf, Whh_bf, b_ih, b_hh, hb[cur], hf[cur], hb[nxt],
            hf[nxt], Wout_bf, b_out, out, t);
        cur = nxt;
    }

    // 4) epilogue: y for the last timestep
    gru_out_kernel<<<dim3(II / 64, BB / 32), 256, 0, stream>>>(
        hb[cur], Wout_bf, b_out, out, TT - 1);
}
